// RecConvLayer_28114855920139
// MI455X (gfx1250) — compile-verified, matrix operand order checked by reading, not run on hardware
//
#include <hip/hip_runtime.h>
#include <hip/hip_bf16.h>

typedef _Float16 h8v  __attribute__((ext_vector_type(8)));
typedef _Float16 v16h __attribute__((ext_vector_type(16)));
typedef float    v8f  __attribute__((ext_vector_type(8)));

#define TSEQ 2304
#define HWPX 9216
#define EPSN 1e-5f

// Assemble a 16-wide f16 WMMA fragment from two 16B loads.
// Layout (16-bit A/B, wave32): lane&15 = row(M)/col(N); half-wave adds K+8;
// VGPR0..3 hold K=base+0..7, VGPR4..7 hold K=base+16..23.
__device__ __forceinline__ v16h frag16(const _Float16* p) {
  h8v lo = *(const h8v*)p;
  h8v hi = *(const h8v*)(p + 16);
  v16h r;
#pragma unroll
  for (int i = 0; i < 8; i++) { r[i] = lo[i]; r[8 + i] = hi[i]; }
  return r;
}

// ---------------------------------------------------------------------------
// Generic WMMA GEMM: out[M,N] = A[M,K](f16) * W[N,K](f16)^T + bias[N], opt
// PReLU, f32 or f16 store. One wave = one 16x16 tile. Batched over blockIdx.z.
// ---------------------------------------------------------------------------
__global__ __launch_bounds__(256) void wmma_gemm_f16(
    const _Float16* __restrict__ A, const _Float16* __restrict__ W,
    const float* __restrict__ bias, const float* __restrict__ alpha,
    void* __restrict__ outp, int M, int N, int K,
    long long aStride, int zDivA, long long wStride, long long bStride,
    long long oStride, int halfOut) {
  int z = blockIdx.z;
  A += (long long)(z / zDivA) * aStride;
  W += (long long)z * wStride;
  if (bias) bias += (long long)z * bStride;
  long long obase = (long long)z * oStride;

  int wv = threadIdx.x >> 5, ln = threadIdx.x & 31;
  int nt = blockIdx.x * 8 + wv;
  if (nt * 16 >= N) return;  // wave-uniform
  int l16 = ln & 15;
  int h8 = (ln >> 4) * 8;
  const _Float16* ar = A + (long long)(blockIdx.y * 16 + l16) * K + h8;
  const _Float16* wr = W + (long long)(nt * 16 + l16) * K + h8;

  v8f acc = {};
  for (int kb = 0; kb < K; kb += 32) {
    v16h a = frag16(ar + kb);
    v16h b = frag16(wr + kb);
    acc = __builtin_amdgcn_wmma_f32_16x16x32_f16(false, a, false, b, (short)0,
                                                 acc, false, false);
  }
  int col = nt * 16 + l16;
  float bc = bias ? bias[col] : 0.f;
  float al = alpha ? alpha[0] : 0.f;
  int r0 = blockIdx.y * 16 + (ln >> 4) * 8;  // C/D layout: VGPR r -> M = r + 8*half
#pragma unroll
  for (int r = 0; r < 8; r++) {
    float v = acc[r] + bc;
    if (alpha) v = v > 0.f ? v : al * v;
    long long idx = obase + (long long)(r0 + r) * N + col;
    if (halfOut) ((_Float16*)outp)[idx] = (_Float16)v;
    else         ((float*)outp)[idx] = v;
  }
}

// ---------------------------------------------------------------------------
// GRU recurrent scan, WMMA version. One block per (gru,group,dir); 24 waves
// each own one 16x16 N-tile of gh = h(16x128,f16) @ whh^T(128x384,f16). whh
// lives in per-wave VGPR fragments; h lives in LDS (f16, padded rows).
// gx (f16, bih folded in) is async-prefetched into double-buffered LDS by
// waves 16..21 (global_load_async_to_lds_b128 + s_wait_asynccnt).
// dir = blockIdx&1 walks t backwards.
// ---------------------------------------------------------------------------
__global__ __launch_bounds__(768) void gru_scan(
    const _Float16* __restrict__ gx, const _Float16* __restrict__ whh,
    const float* __restrict__ bhh, float* __restrict__ yout, int T) {
  int gd = blockIdx.x;
  gx += (long long)gd * T * 4 * 384;
  whh += (long long)gd * 384 * 128;
  bhh += gd * 384;
  yout += (long long)gd * T * 4 * 128;
  int dir = gd & 1;

  __shared__ _Float16 hbuf[16][136];   // padded f16 h (rows 4..15 stay zero)
  __shared__ float ghs[4][384];
  __shared__ _Float16 gxb[2][1536];    // double-buffered gx[t] staging

  int tid = threadIdx.x, wv = tid >> 5, ln = tid & 31;
  for (int i = tid; i < 16 * 136; i += 768) (&hbuf[0][0])[i] = (_Float16)0.f;

  bool issuer = (tid >= 512 && tid < 704);           // waves 16..21, 192 lanes
  unsigned voff = (unsigned)(tid - 512) * 16u;       // byte offset in 3KB chunk
  auto issue = [&](int t, int buf) {
    unsigned long long sb = (unsigned long long)(uintptr_t)(gx + (size_t)t * 1536);
    unsigned dst = (unsigned)(uintptr_t)(const void*)(&gxb[buf][0]) + voff;
    asm volatile("global_load_async_to_lds_b128 %0, %1, %2"
                 :: "v"(dst), "v"(voff), "s"(sb) : "memory");
  };
  if (issuer) issue(dir ? T - 1 : 0, 0);

  int l16 = ln & 15, h8 = (ln >> 4) * 8;
  v16h bf[4];  // whh fragments, resident for whole scan
  {
    const _Float16* wr = whh + (long long)(wv * 16 + l16) * 128 + h8;
#pragma unroll
    for (int kt = 0; kt < 4; kt++) bf[kt] = frag16(wr + kt * 32);
  }
  int gb = tid >> 7, gc = tid & 127;  // gate threads: tid<512 -> (batch, chan)
  float hreg = 0.f, br = 0.f, bz = 0.f, bn = 0.f;
  if (tid < 512) { br = bhh[gc]; bz = bhh[128 + gc]; bn = bhh[256 + gc]; }
  if (issuer) asm volatile("s_wait_asynccnt 0" ::: "memory");
  __syncthreads();

  for (int s = 0; s < T; s++) {
    int t = dir ? (T - 1 - s) : s;
    v8f acc = {};
#pragma unroll
    for (int kt = 0; kt < 4; kt++) {
      v16h a = frag16(&hbuf[l16][kt * 32 + h8]);
      acc = __builtin_amdgcn_wmma_f32_16x16x32_f16(false, a, false, bf[kt],
                                                   (short)0, acc, false, false);
    }
    if (ln < 16) {  // rows 0..3 (batch) live in VGPRs 0..3 of lanes 0..15
#pragma unroll
      for (int r = 0; r < 4; r++) ghs[r][wv * 16 + ln] = acc[r];
    }
    __syncthreads();
    if (tid < 512) {
      const _Float16* gq = &gxb[s & 1][gb * 384 + gc];
      float ur = (float)gq[0]   + ghs[gb][gc]       + br;
      float uz = (float)gq[128] + ghs[gb][128 + gc] + bz;
      float rr = 1.f / (1.f + __expf(-ur));
      float zz = 1.f / (1.f + __expf(-uz));
      float un = (float)gq[256] + rr * (ghs[gb][256 + gc] + bn);
      float e2 = __expf(-2.f * un);
      float nn = (1.f - e2) / (1.f + e2);
      hreg = (1.f - zz) * nn + zz * hreg;
      yout[((long long)t * 4 + gb) * 128 + gc] = hreg;
      hbuf[gb][gc] = (_Float16)hreg;
    } else if (issuer && s + 1 < T) {
      issue(dir ? T - 2 - s : s + 1, (s + 1) & 1);
    }
    if (issuer) asm volatile("s_wait_asynccnt 0" ::: "memory");
    __syncthreads();
  }
}

// ---------------------------------------------------------------------------
// Direct 3x3 conv (pad 1) + bias + PReLU, fp32. Cout=128 fixed.
// ---------------------------------------------------------------------------
__global__ __launch_bounds__(256) void conv3x3_prelu(
    const float* __restrict__ in, const float* __restrict__ wgt,
    const float* __restrict__ bias, const float* __restrict__ alpha,
    float* __restrict__ out, int Cin) {
  __shared__ float wsm[1152];
  int co = blockIdx.y, b = blockIdx.z;
  int nw = Cin * 9;
  for (int i = threadIdx.x; i < nw; i += 256) wsm[i] = wgt[co * nw + i];
  __syncthreads();
  int pix = blockIdx.x * 256 + threadIdx.x;
  int y = pix / 96, x = pix % 96;
  float acc = bias[co];
  for (int ci = 0; ci < Cin; ci++) {
    const float* ip = in + ((long long)(b * Cin + ci)) * HWPX;
    const float* wp = wsm + ci * 9;
#pragma unroll
    for (int ky = 0; ky < 3; ky++) {
      int yy = y + ky - 1;
      if (yy < 0 || yy > 95) continue;
#pragma unroll
      for (int kx = 0; kx < 3; kx++) {
        int xx = x + kx - 1;
        if (xx < 0 || xx > 95) continue;
        acc += ip[yy * 96 + xx] * wp[ky * 3 + kx];
      }
    }
  }
  float al = alpha[0];
  acc = acc > 0.f ? acc : al * acc;
  out[((long long)(b * 128 + co)) * HWPX + pix] = acc;
}

// ---- GroupNorm (NCHW layout, 8 groups of 16 channels) ----
__global__ __launch_bounds__(256) void gn_stats_nchw(const float* __restrict__ x,
                                                     float* __restrict__ stats) {
  int b = blockIdx.x >> 3, g = blockIdx.x & 7;
  const float* base = x + ((long long)(b * 128 + g * 16)) * HWPX;  // contiguous
  const int n = 16 * HWPX;
  __shared__ float s1[256], s2[256];
  float a = 0.f, q = 0.f;
  for (int i = threadIdx.x; i < n; i += 256) { float v = base[i]; a += v; q += v * v; }
  s1[threadIdx.x] = a; s2[threadIdx.x] = q;
  __syncthreads();
  for (int off = 128; off; off >>= 1) {
    if (threadIdx.x < off) { s1[threadIdx.x] += s1[threadIdx.x + off]; s2[threadIdx.x] += s2[threadIdx.x + off]; }
    __syncthreads();
  }
  if (!threadIdx.x) {
    float mean = s1[0] / n;
    float var = s2[0] / n - mean * mean;
    stats[blockIdx.x * 2] = mean;
    stats[blockIdx.x * 2 + 1] = rsqrtf(var + EPSN);
  }
}

__global__ void gn_apply_nchw(float* __restrict__ x, const float* __restrict__ stats,
                              const float* __restrict__ gw, const float* __restrict__ gb) {
  long long i = (long long)blockIdx.x * 256 + threadIdx.x;
  if (i >= (long long)4 * 128 * HWPX) return;
  int c = (int)((i / HWPX) % 128);
  int b = (int)(i / ((long long)128 * HWPX));
  int g = c >> 4;
  float m = stats[(b * 8 + g) * 2], r = stats[(b * 8 + g) * 2 + 1];
  x[i] = (x[i] - m) * r * gw[c] + gb[c];
}

// ---- GroupNorm for row-major (pix,128) combine output, + transpose to NCHW ----
__global__ __launch_bounds__(256) void gn_stats_rm(const float* __restrict__ x,
                                                   float* __restrict__ stats) {
  int b = blockIdx.x >> 3, g = blockIdx.x & 7;
  const int n = 16 * HWPX;
  __shared__ float s1[256], s2[256];
  float a = 0.f, q = 0.f;
  for (int i = threadIdx.x; i < n; i += 256) {
    int rr = i >> 4, cc = i & 15;
    float v = x[((long long)(b * HWPX + rr)) * 128 + g * 16 + cc];
    a += v; q += v * v;
  }
  s1[threadIdx.x] = a; s2[threadIdx.x] = q;
  __syncthreads();
  for (int off = 128; off; off >>= 1) {
    if (threadIdx.x < off) { s1[threadIdx.x] += s1[threadIdx.x + off]; s2[threadIdx.x] += s2[threadIdx.x + off]; }
    __syncthreads();
  }
  if (!threadIdx.x) {
    float mean = s1[0] / n;
    float var = s2[0] / n - mean * mean;
    stats[blockIdx.x * 2] = mean;
    stats[blockIdx.x * 2 + 1] = rsqrtf(var + EPSN);
  }
}

__global__ void gn_apply_rm_t(const float* __restrict__ x, const float* __restrict__ stats,
                              const float* __restrict__ gw, const float* __restrict__ gb,
                              float* __restrict__ out) {
  long long i = (long long)blockIdx.x * 256 + threadIdx.x;
  if (i >= (long long)4 * 128 * HWPX) return;
  int c = (int)(i % 128);
  int pix = (int)((i / 128) % HWPX);
  int b = (int)(i / ((long long)128 * HWPX));
  int g = c >> 4;
  float m = stats[(b * 8 + g) * 2], r = stats[(b * 8 + g) * 2 + 1];
  float v = (x[i] - m) * r * gw[c] + gb[c];
  out[((long long)(b * 128 + c)) * HWPX + pix] = v;
}

// ---- elementwise / gather helpers ----
__global__ void cvt_f16(const float* __restrict__ in, _Float16* __restrict__ out, long long n) {
  long long i = (long long)blockIdx.x * 256 + threadIdx.x;
  if (i < n) out[i] = (_Float16)in[i];
}

// seq[g][t][b][c] (f16) <- x2 NCHW. vertical: row=t%96,col=g*24+t/96; else swapped.
__global__ void gather_seq(const float* __restrict__ x, _Float16* __restrict__ out, int vertical) {
  int i = blockIdx.x * 256 + threadIdx.x;
  if (i >= 4718592) return;
  int c = i & 127, b = (i >> 7) & 3, t = (i >> 9) % TSEQ, g = i / (TSEQ * 512);
  int row, col;
  if (vertical) { row = t % 96; col = g * 24 + t / 96; }
  else          { row = g * 24 + t / 96; col = t % 96; }
  out[i] = (_Float16)x[((long long)(b * 128 + c)) * HWPX + row * 96 + col];
}

// Resequence between H-format and V-format seq tensors (f32 src -> f16 dst).
__global__ void reseq(const float* __restrict__ src, _Float16* __restrict__ dst, int h2v) {
  int i = blockIdx.x * 256 + threadIdx.x;
  if (i >= 4718592) return;
  int c = i & 127, b = (i >> 7) & 3, s = (i >> 9) % TSEQ, g = i / (TSEQ * 512);
  int row, col, g1, t1;
  if (h2v) {  // dst is V-format, src is H-format
    row = s % 96; col = g * 24 + s / 96;
    g1 = row / 24; t1 = (row % 24) * 96 + col;
  } else {    // dst is H-format, src is V-format
    row = g * 24 + s / 96; col = s % 96;
    g1 = col / 24; t1 = (col % 24) * 96 + row;
  }
  dst[i] = (_Float16)src[(((long long)g1 * TSEQ + t1) * 4 + b) * 128 + c];
}

// in16[G8][t][b][256] <- concat dirs of y[G8][2][t][b][128]   (G8 = 2 grus * 4)
__global__ void concat_dirs(const float* __restrict__ y, _Float16* __restrict__ out) {
  int i = blockIdx.x * 256 + threadIdx.x;
  if (i >= 18874368) return;
  int c2 = i & 255, b = (i >> 8) & 3, t = (i >> 10) % TSEQ, g = i / (TSEQ * 1024);
  int d = c2 >> 7, c = c2 & 127;
  out[i] = (_Float16)y[((((long long)g * 2 + d) * TSEQ + t) * 4 + b) * 128 + c];
}

// hn[G8][t][b][c] = mean over dirs of y[G8][2][t][b][c]
__global__ void mean_dirs(const float* __restrict__ y, float* __restrict__ out) {
  int i = blockIdx.x * 256 + threadIdx.x;
  if (i >= 9437184) return;
  int c = i & 127, b = (i >> 7) & 3, t = (i >> 9) % TSEQ, g = i / (TSEQ * 512);
  long long f = (((long long)g * 2 * TSEQ + t) * 4 + b) * 128 + c;
  long long ds = (long long)TSEQ * 4 * 128;
  out[i] = 0.5f * (y[f] + y[f + ds]);
}

// comb_in16[b][pix][256]: ch<128 from hn2 (H-format), ch>=128 from vn2 (V-format)
__global__ void gather_comb(const float* __restrict__ hn2, const float* __restrict__ vn2,
                            _Float16* __restrict__ out) {
  int i = blockIdx.x * 256 + threadIdx.x;
  if (i >= 9437184) return;
  int c2 = i & 255, pix = (i >> 8) % HWPX, b = i / (HWPX * 256);
  int row = pix / 96, col = pix % 96;
  float v;
  if (c2 < 128) {
    int g = row / 24, t = (row % 24) * 96 + col;
    v = hn2[(((long long)g * TSEQ + t) * 4 + b) * 128 + c2];
  } else {
    int c = c2 - 128, g = col / 24, s = (col % 24) * 96 + row;
    v = vn2[(((long long)g * TSEQ + s) * 4 + b) * 128 + c];
  }
  out[i] = (_Float16)v;
}

// ---------------------------------------------------------------------------
extern "C" void kernel_launch(void* const* d_in, const int* in_sizes, int n_in,
                              void* d_out, int out_size, void* d_ws, size_t ws_size,
                              hipStream_t stream) {
  (void)in_sizes; (void)n_in; (void)out_size; (void)ws_size;
  const float* x       = (const float*)d_in[0];
  const float* conv1_w = (const float*)d_in[1];
  const float* conv1_b = (const float*)d_in[2];
  const float* prelu1  = (const float*)d_in[3];
  const float* gn1_w   = (const float*)d_in[4];
  const float* gn1_b   = (const float*)d_in[5];
  const float* conv2_w = (const float*)d_in[6];
  const float* conv2_b = (const float*)d_in[7];
  const float* prelu2  = (const float*)d_in[8];
  const float* gn2_w   = (const float*)d_in[9];
  const float* gn2_b   = (const float*)d_in[10];
  const float* comb_w  = (const float*)d_in[11];
  const float* comb_b  = (const float*)d_in[12];
  const float* prelu3  = (const float*)d_in[13];
  const float* gn3_w   = (const float*)d_in[14];
  const float* gn3_b   = (const float*)d_in[15];
  const float *wih0[4], *whh0[4], *bih0[4], *bhh0[4];
  const float *wih1[4], *whh1[4], *bih1[4], *bhh1[4];
  for (int p = 0; p < 4; p++) {
    int base = 16 + 8 * p;  // order: h, v, h1, v1
    wih0[p] = (const float*)d_in[base + 0];
    whh0[p] = (const float*)d_in[base + 1];
    bih0[p] = (const float*)d_in[base + 2];
    bhh0[p] = (const float*)d_in[base + 3];
    wih1[p] = (const float*)d_in[base + 4];
    whh1[p] = (const float*)d_in[base + 5];
    bih1[p] = (const float*)d_in[base + 6];
    bhh1[p] = (const float*)d_in[base + 7];
  }

  // ---- workspace carve ----
  char* wp = (char*)d_ws;
  auto carve = [&](size_t bytes) { void* p = wp; wp += (bytes + 255) & ~(size_t)255; return p; };
  const long long NX = 4LL * 128 * HWPX;  // 4,718,592
  float*     x1      = (float*)carve(NX * 4);
  float*     x2      = (float*)carve(NX * 4);
  _Float16*  seqA16  = (_Float16*)carve(NX * 2);  // H seqs / y2b  (adjacent to seqB16)
  _Float16*  seqB16  = (_Float16*)carve(NX * 2);  // V seqs / y1b
  _Float16*  wih0_16 = (_Float16*)carve(4LL * 393216 * 2);
  _Float16*  wih1_16 = (_Float16*)carve(4LL * 786432 * 2);
  _Float16*  whh0_16 = (_Float16*)carve(4LL * 393216 * 2);
  _Float16*  whh1_16 = (_Float16*)carve(4LL * 393216 * 2);
  _Float16*  combw16 = (_Float16*)carve(32768 * 2);
  _Float16*  gx16    = (_Float16*)carve(16LL * TSEQ * 4 * 384 * 2);  // also comb bufs
  float*     ybuf    = (float*)carve(16LL * TSEQ * 4 * 128 * 4);
  _Float16*  in16    = (_Float16*)carve(8LL * TSEQ * 4 * 256 * 2);
  float*     hn      = (float*)carve(NX * 4);  // adjacent to vn; reused for hn2
  float*     vn      = (float*)carve(NX * 4);  // reused for vn2
  float*     bih0c   = (float*)carve(4LL * 3072 * 4);
  float*     bhh0c   = (float*)carve(4LL * 3072 * 4);
  float*     bih1c   = (float*)carve(4LL * 3072 * 4);
  float*     bhh1c   = (float*)carve(4LL * 3072 * 4);
  float*     stats   = (float*)carve(64 * 4);
  _Float16*  comb_in16 = (_Float16*)gx16;  // alias (gx dead by then)
  float*     comb_out  = (float*)((char*)gx16 + ((NX * 2 + 255) & ~(size_t)255));

  int eg = (int)((NX + 255) / 256);  // 18432

  // ---- weight conversion to f16 + contiguous bias pools ----
  for (int p = 0; p < 4; p++) {
    cvt_f16<<<1536, 256, 0, stream>>>(wih0[p], wih0_16 + (long long)p * 393216, 393216);
    cvt_f16<<<3072, 256, 0, stream>>>(wih1[p], wih1_16 + (long long)p * 786432, 786432);
    cvt_f16<<<1536, 256, 0, stream>>>(whh0[p], whh0_16 + (long long)p * 393216, 393216);
    cvt_f16<<<1536, 256, 0, stream>>>(whh1[p], whh1_16 + (long long)p * 393216, 393216);
    hipMemcpyAsync(bih0c + p * 3072, bih0[p], 3072 * 4, hipMemcpyDeviceToDevice, stream);
    hipMemcpyAsync(bhh0c + p * 3072, bhh0[p], 3072 * 4, hipMemcpyDeviceToDevice, stream);
    hipMemcpyAsync(bih1c + p * 3072, bih1[p], 3072 * 4, hipMemcpyDeviceToDevice, stream);
    hipMemcpyAsync(bhh1c + p * 3072, bhh1[p], 3072 * 4, hipMemcpyDeviceToDevice, stream);
  }
  cvt_f16<<<128, 256, 0, stream>>>(comb_w, combw16, 32768);

  // ---- conv1 -> PReLU -> GN ----
  conv3x3_prelu<<<dim3(36, 128, 4), 256, 0, stream>>>(x, conv1_w, conv1_b, prelu1, x1, 3);
  gn_stats_nchw<<<32, 256, 0, stream>>>(x1, stats);
  gn_apply_nchw<<<eg, 256, 0, stream>>>(x1, stats, gn1_w, gn1_b);

  // ---- conv2 -> PReLU -> GN ----
  conv3x3_prelu<<<dim3(36, 128, 4), 256, 0, stream>>>(x1, conv2_w, conv2_b, prelu2, x2, 128);
  gn_stats_nchw<<<32, 256, 0, stream>>>(x2, stats);
  gn_apply_nchw<<<eg, 256, 0, stream>>>(x2, stats, gn2_w, gn2_b);

  // ---- build sequence inputs (seqA16/seqB16 contiguous: 8 batched slots) ----
  gather_seq<<<eg, 256, 0, stream>>>(x2, seqA16, 0);  // horizontal
  gather_seq<<<eg, 256, 0, stream>>>(x2, seqB16, 1);  // vertical

  // ---- paired bidirectional 2-layer GRU stacks (h+v, then h1+v1) ----
  auto run_pair = [&](int P, float* hout) {
    dim3 gg(3, 576, 16);
    // layer0: gx = seq @ wih0^T + bih0   (K=128), f16 out
    wmma_gemm_f16<<<gg, 256, 0, stream>>>(
        seqA16, wih0_16 + (long long)P * 2 * 393216, bih0c + P * 6144, nullptr,
        gx16, 9216, 384, 128, (long long)TSEQ * 4 * 128, 2,
        49152LL, 384LL, 9216LL * 384, 1);
    gru_scan<<<16, 768, 0, stream>>>(gx16, whh0_16 + (long long)P * 2 * 393216,
                                     bhh0c + P * 6144, ybuf, TSEQ);
    concat_dirs<<<73728, 256, 0, stream>>>(ybuf, in16);
    // layer1: K=256
    wmma_gemm_f16<<<gg, 256, 0, stream>>>(
        in16, wih1_16 + (long long)P * 2 * 786432, bih1c + P * 6144, nullptr,
        gx16, 9216, 384, 256, (long long)TSEQ * 4 * 256, 2,
        98304LL, 384LL, 9216LL * 384, 1);
    gru_scan<<<16, 768, 0, stream>>>(gx16, whh1_16 + (long long)P * 2 * 393216,
                                     bhh1c + P * 6144, ybuf, TSEQ);
    mean_dirs<<<36864, 256, 0, stream>>>(ybuf, hout);  // writes hn then vn
  };

  run_pair(0, hn);  // h -> hn, v -> vn

  // resequence for round 2 (overwrite seq buffers)
  reseq<<<eg, 256, 0, stream>>>(vn, seqA16, 0);  // V-format vn -> H-format y2b
  reseq<<<eg, 256, 0, stream>>>(hn, seqB16, 1);  // H-format hn -> V-format y1b

  run_pair(1, hn);  // h1 -> hn2 (in hn), v1 -> vn2 (in vn)

  // ---- combine conv (1x1 as GEMM) -> PReLU -> GN -> NCHW output ----
  gather_comb<<<36864, 256, 0, stream>>>(hn, vn, comb_in16);
  wmma_gemm_f16<<<dim3(1, 2304, 1), 256, 0, stream>>>(
      comb_in16, combw16, comb_b, prelu3, comb_out,
      36864, 128, 256, 0, 1, 0, 0, 0, 0);
  gn_stats_rm<<<32, 256, 0, stream>>>(comb_out, stats);
  gn_apply_rm_t<<<eg, 256, 0, stream>>>(comb_out, stats, gn3_w, gn3_b, (float*)d_out);
}
